// WilliamsAttention_2362232013183
// MI455X (gfx1250) — matrix operand order repaired and checked by reading, not compiled, not run on hardware
//
#include <hip/hip_runtime.h>
#include <stdint.h>

typedef __attribute__((ext_vector_type(16))) _Float16 v16h;
typedef __attribute__((ext_vector_type(8)))  _Float16 v8h;
typedef __attribute__((ext_vector_type(8)))  float    v8f;
typedef int8_t v8i8 __attribute__((ext_vector_type(8)));

union frag16 { v16h v; v8h h[2]; };

// Problem constants (match reference)
constexpr int Cdim = 1024;   // channels
constexpr int Tdim = 2048;   // sequence
constexpr int Hn   = 16;     // heads
constexpr int Dh   = 64;     // head dim
constexpr int Mrows = 2 * Tdim;            // B*T = 4096
#define ATTN_SCALE 0.125f                  // 1/sqrt(64)

#if __has_builtin(__builtin_amdgcn_global_load_async_to_lds_b128) && \
    __has_builtin(__builtin_amdgcn_s_wait_asynccnt)
#define CDNA5_ASYNC 1
#define AS1 __attribute__((address_space(1)))
#define AS3 __attribute__((address_space(3)))
// Builtin prototype (from clang diagnostic):
//   void (v4i32 AS1* gaddr, v4i32 AS3* lds, imm int offset, imm int cpol)
typedef int v4i32 __attribute__((vector_size(16)));
#endif

// ---------------------------------------------------------------------------
// GEMM: out[M=4096][N=1024] = f16(X) @ f16(dequant(W)).T + bias
//   MODE 0: OutF f32 row-major [M][N]               -- final projection
//   MODE 1: OutH f16 scattered to [B,H,T,D]         -- q / k
//   MODE 2: OutH f16 scattered to [B,H,D,T] (v^T)   -- v transposed
// 256 threads (8 waves). Block tile 128x64, K-step 32, double-buffered LDS,
// software pipelined: next chunk's global loads issue before current WMMAs.
// Wave w -> 32x32 tile: 4 accumulators, 4 WMMA / k-step.
// ---------------------------------------------------------------------------
template <int MODE>
__global__ __launch_bounds__(256)
void gemm_i8(const float* __restrict__ X, const int8_t* __restrict__ W,
             const float* __restrict__ S, const float* __restrict__ Bias,
             _Float16* __restrict__ OutH, float* __restrict__ OutF) {
  constexpr int LDA = 40;  // halfs (80B, 16B aligned, bank-skewed)
  constexpr int LDB = 40;
  __shared__ __align__(16) _Float16 As[2][128 * LDA];
  __shared__ __align__(16) _Float16 Bs[2][64 * LDB];

  const int tid  = threadIdx.x;
  const int wave = tid >> 5, lane = tid & 31;
  const int m0 = blockIdx.y * 128;
  const int n0 = blockIdx.x * 64;
  const int wm = (wave >> 1) * 32;
  const int wn = (wave & 1) * 32;
  const int l15 = lane & 15;
  const bool lo = (lane < 16);

  // staging assignments
  const int ar = tid >> 1, ac = (tid & 1) * 16;   // A: 128 rows x 32 f32
  const int br = tid >> 2, bc = (tid & 3) * 8;    // B: 64 rows x 32 int8
  const float bscale = S[n0 + br];
  const float*  xbase = X + (size_t)(m0 + ar) * Cdim + ac;
  const int8_t* wbase = W + (size_t)(n0 + br) * Cdim + bc;

  float fx[16];
  v8i8  wb;
  auto gload = [&](int k0) {
    const float* xp = xbase + k0;
    *(float4*)&fx[0]  = *(const float4*)(xp + 0);
    *(float4*)&fx[4]  = *(const float4*)(xp + 4);
    *(float4*)&fx[8]  = *(const float4*)(xp + 8);
    *(float4*)&fx[12] = *(const float4*)(xp + 12);
    wb = *(const v8i8*)(wbase + k0);
  };
  auto commit = [&](int buf) {
    v8h a0, a1, hb;
#pragma unroll
    for (int i = 0; i < 8; ++i) {
      a0[i] = (_Float16)fx[i];
      a1[i] = (_Float16)fx[8 + i];
      hb[i] = (_Float16)((float)wb[i] * bscale);
    }
    *(v8h*)&As[buf][ar * LDA + ac]     = a0;
    *(v8h*)&As[buf][ar * LDA + ac + 8] = a1;
    *(v8h*)&Bs[buf][br * LDB + bc]     = hb;
  };

  gload(0);
  commit(0);
  int p = 0;
  v8f acc[2][2] = {};

  for (int k0 = 0; k0 < Cdim; k0 += 32) {
    __syncthreads();
    const bool more = (k0 + 32 < Cdim);
    if (more) gload(k0 + 32);            // in flight during the WMMAs below

    frag16 af[2], bf[2];
#pragma unroll
    for (int mt = 0; mt < 2; ++mt) {
      const int arow = wm + mt * 16 + l15;
      const int klo = lo ? 0 : 8;        // A layout: {0..7,16..23}/{8..15,24..31}
      af[mt].h[0] = *(const v8h*)&As[p][arow * LDA + klo];
      af[mt].h[1] = *(const v8h*)&As[p][arow * LDA + klo + 16];
    }
#pragma unroll
    for (int nt = 0; nt < 2; ++nt) {
      const int bcol = wn + nt * 16 + l15;
      const int blo = lo ? 0 : 8;        // B layout: {0..7,16..23}/{8..15,24..31}
      bf[nt].h[0] = *(const v8h*)&Bs[p][bcol * LDB + blo];
      bf[nt].h[1] = *(const v8h*)&Bs[p][bcol * LDB + blo + 16];
    }
#pragma unroll
    for (int mt = 0; mt < 2; ++mt)
#pragma unroll
      for (int nt = 0; nt < 2; ++nt)
        acc[mt][nt] = __builtin_amdgcn_wmma_f32_16x16x32_f16(
            false, af[mt].v, false, bf[nt].v, (short)0, acc[mt][nt], false, false);

    if (more) commit(p ^ 1);             // loadcnt wait lands here, after WMMAs
    p ^= 1;
  }

  // ---- epilogue (compile-time store path) ----
#pragma unroll
  for (int mt = 0; mt < 2; ++mt)
#pragma unroll
    for (int nt = 0; nt < 2; ++nt) {
      const int col  = n0 + wn + nt * 16 + l15;
      const float bias = Bias[col];
      const int row0 = m0 + wm + mt * 16 + (lo ? 0 : 8);
      if constexpr (MODE == 0) {
        float* op = OutF + (size_t)row0 * Cdim + col;
#pragma unroll
        for (int i = 0; i < 8; ++i) op[(size_t)i * Cdim] = acc[mt][nt][i] + bias;
      } else if constexpr (MODE == 1) {
        const int b = row0 >> 11, t0 = row0 & (Tdim - 1);
        const int hh = col >> 6, d = col & 63;
        _Float16* op = OutH + ((size_t)(b * Hn + hh) * Tdim + t0) * Dh + d;
#pragma unroll
        for (int i = 0; i < 8; ++i) op[i * Dh] = (_Float16)(acc[mt][nt][i] + bias);
      } else {
        const int b = row0 >> 11, t0 = row0 & (Tdim - 1);
        const int hh = col >> 6, d = col & 63;
        v8h pk;
#pragma unroll
        for (int i = 0; i < 8; ++i) pk[i] = (_Float16)(acc[mt][nt][i] + bias);
        // 8 consecutive t positions -> one 16B store
        *(v8h*)(OutH + ((size_t)(b * Hn + hh) * Dh + d) * Tdim + t0) = pk;
      }
    }
}

// ---------------------------------------------------------------------------
// Causal flash attention. Q,K: f16 [B,H,T,D]; VT: f16 [B,H,D,T]; Y: f32 [B,T,C].
// Grid (T/128, B*H). Block 256 = 8 waves; wave w owns queries qt*128+w*16 ..+15.
// Per 32-key chunk (K/V staged to LDS with GLOBAL_LOAD_ASYNC_TO_LDS_B128):
//   S^T = K@Q^T (4 wmma), online softmax (lane = query column, per-lane reduce
//   + one shfl_xor 16), P->per-wave LDS [m][j], O^T += V^T@P^T (4 wmma).
// ---------------------------------------------------------------------------
__global__ __launch_bounds__(256)
void flash_attn(const _Float16* __restrict__ Q, const _Float16* __restrict__ K,
                const _Float16* __restrict__ VT, float* __restrict__ Y) {
  constexpr int LDK = 72;  // halfs, [j=32][d=64]
  constexpr int LDV = 40;  // halfs, [d=64][j=32]
  constexpr int LDP = 40;  // halfs, [m=16][j=32] per wave
  __shared__ __align__(16) _Float16 Ks[32 * LDK];
  __shared__ __align__(16) _Float16 Vs[64 * LDV];
  __shared__ __align__(16) _Float16 Ps[8][16 * LDP];

  const int bh = blockIdx.y;                 // b*16 + h
  const int qt = blockIdx.x;
  const _Float16* Qp = Q + (size_t)bh * Tdim * Dh;
  const _Float16* Kp = K + (size_t)bh * Tdim * Dh;
  const _Float16* Vp = VT + (size_t)bh * Dh * Tdim;

  const int tid = threadIdx.x;
  const int wave = tid >> 5, lane = tid & 31;
  const int l15 = lane & 15;
  const bool lolane = (lane < 16);
  const int qlo = qt * 128 + wave * 16;
  const int qrow = qlo + l15;                // this lane's query column

  // staging assignments
  const int kr = tid >> 3, kcl = (tid & 7) * 8;   // K: 32 x 64 halfs
  const int vr = tid >> 2, vcl = (tid & 3) * 8;   // VT: 64 x 32 halfs

  // Q fragments (B-layout for Q^T), pre-scaled by 1/sqrt(D)
  frag16 qf[2];
#pragma unroll
  for (int s = 0; s < 2; ++s) {
    const int dlo = s * 32 + (lolane ? 0 : 8);
    const _Float16* qp = Qp + (size_t)qrow * Dh + dlo;
    qf[s].h[0] = *(const v8h*)qp;
    qf[s].h[1] = *(const v8h*)(qp + 16);
#pragma unroll
    for (int i = 0; i < 16; ++i)
      qf[s].v[i] = (_Float16)((float)qf[s].v[i] * ATTN_SCALE);
  }

  float mrow = -1e30f, lrow = 0.f;
  v8f ot[4] = {};

  const int kend = qt * 128 + 128;           // block's causal key bound
  for (int kc = 0; kc < kend; kc += 32) {
    // ---- stage K chunk [32][64], V^T chunk [64][32] into LDS ----
#ifdef CDNA5_ASYNC
    __builtin_amdgcn_global_load_async_to_lds_b128(
        (AS1 v4i32*)(Kp + (size_t)(kc + kr) * Dh + kcl),
        (AS3 v4i32*)(&Ks[kr * LDK + kcl]), 0, 0);
    __builtin_amdgcn_global_load_async_to_lds_b128(
        (AS1 v4i32*)(Vp + (size_t)vr * Tdim + kc + vcl),
        (AS3 v4i32*)(&Vs[vr * LDV + vcl]), 0, 0);
    __builtin_amdgcn_s_wait_asynccnt(0);
#else
    *(v8h*)&Ks[kr * LDK + kcl] = *(const v8h*)(Kp + (size_t)(kc + kr) * Dh + kcl);
    *(v8h*)&Vs[vr * LDV + vcl] = *(const v8h*)(Vp + (size_t)vr * Tdim + kc + vcl);
#endif
    __syncthreads();

    if (kc <= qlo + 15) {                    // wave-uniform: EXEC stays full
      // ---- S^T = K @ Q^T ----
      v8f st[2] = {};
#pragma unroll
      for (int jt = 0; jt < 2; ++jt) {
        const int jrow = jt * 16 + l15;
#pragma unroll
        for (int s = 0; s < 2; ++s) {
          frag16 kf;
          const int dlo = s * 32 + (lolane ? 0 : 8);
          kf.h[0] = *(const v8h*)&Ks[jrow * LDK + dlo];
          kf.h[1] = *(const v8h*)&Ks[jrow * LDK + dlo + 16];
          st[jt] = __builtin_amdgcn_wmma_f32_16x16x32_f16(
              false, kf.v, false, qf[s].v, (short)0, st[jt], false, false);
        }
      }
      // ---- causal mask (diagonal chunk only) ----
      if (kc + 31 > qlo) {
#pragma unroll
        for (int jt = 0; jt < 2; ++jt)
#pragma unroll
          for (int i = 0; i < 8; ++i) {
            const int j = kc + jt * 16 + (lolane ? i : 8 + i);
            if (j > qrow) st[jt][i] = -1e30f;
          }
      }
      // ---- online softmax (lane = query m; j split across half-waves) ----
      float cmax = -1e30f;
#pragma unroll
      for (int i = 0; i < 8; ++i) {
        cmax = fmaxf(cmax, st[0][i]);
        cmax = fmaxf(cmax, st[1][i]);
      }
      cmax = fmaxf(cmax, __shfl_xor(cmax, 16, 32));
      const float mnew = fmaxf(mrow, cmax);
      const float alpha = __expf(mrow - mnew);
      float psum = 0.f;
#pragma unroll
      for (int i = 0; i < 8; ++i) {
        const float p0 = __expf(st[0][i] - mnew);
        const float p1 = __expf(st[1][i] - mnew);
        st[0][i] = p0; st[1][i] = p1;
        psum += p0 + p1;
      }
      psum += __shfl_xor(psum, 16, 32);
      lrow = lrow * alpha + psum;
      mrow = mnew;
#pragma unroll
      for (int dt = 0; dt < 4; ++dt)
#pragma unroll
        for (int i = 0; i < 8; ++i) ot[dt][i] *= alpha;

      // ---- P (f16) -> per-wave LDS tile in [m][j] (B-fragment) layout ----
      v8h ph0, ph1;
#pragma unroll
      for (int i = 0; i < 8; ++i) {
        ph0[i] = (_Float16)st[0][i];
        ph1[i] = (_Float16)st[1][i];
      }
      const int po = l15 * LDP + (lolane ? 0 : 8);
      *(v8h*)&Ps[wave][po]      = ph0;       // j = 0..7  / 8..15
      *(v8h*)&Ps[wave][po + 16] = ph1;       // j = 16..23 / 24..31

      // ---- O^T += V^T @ P^T ----
      frag16 pf;
      const int pb = l15 * LDP + (lolane ? 0 : 8);
      pf.h[0] = *(const v8h*)&Ps[wave][pb];
      pf.h[1] = *(const v8h*)&Ps[wave][pb + 16];
#pragma unroll
      for (int dt = 0; dt < 4; ++dt) {
        frag16 vf;
        const int drow = dt * 16 + l15;
        const int jlo = lolane ? 0 : 8;
        vf.h[0] = *(const v8h*)&Vs[drow * LDV + jlo];
        vf.h[1] = *(const v8h*)&Vs[drow * LDV + jlo + 16];
        ot[dt] = __builtin_amdgcn_wmma_f32_16x16x32_f16(
            false, vf.v, false, pf.v, (short)0, ot[dt], false, false);
      }
    }
    __syncthreads();
  }

  // ---- normalize + store y[b][t][h*64+d] (f32) ----
  const float inv = 1.f / (lrow + 1e-8f);
  const int bb = bh >> 4, hh = bh & 15;
  const size_t base = ((size_t)bb * Tdim + qrow) * Cdim + hh * Dh;
#pragma unroll
  for (int dt = 0; dt < 4; ++dt) {
    const int d0 = dt * 16 + (lolane ? 0 : 8);
    float4 o0 = make_float4(ot[dt][0] * inv, ot[dt][1] * inv, ot[dt][2] * inv, ot[dt][3] * inv);
    float4 o1 = make_float4(ot[dt][4] * inv, ot[dt][5] * inv, ot[dt][6] * inv, ot[dt][7] * inv);
    *(float4*)&Y[base + d0]     = o0;
    *(float4*)&Y[base + d0 + 4] = o1;
  }
}

// ---------------------------------------------------------------------------
extern "C" void kernel_launch(void* const* d_in, const int* in_sizes, int n_in,
                              void* d_out, int out_size, void* d_ws, size_t ws_size,
                              hipStream_t stream) {
  (void)in_sizes; (void)n_in; (void)out_size; (void)ws_size;
  const float*  x  = (const float*)d_in[0];
  const int8_t* wq = (const int8_t*)d_in[1];
  const float*  sq = (const float*)d_in[2];
  const float*  bq = (const float*)d_in[3];
  const int8_t* wk = (const int8_t*)d_in[4];
  const float*  sk = (const float*)d_in[5];
  const float*  bk = (const float*)d_in[6];
  const int8_t* wv = (const int8_t*)d_in[7];
  const float*  sv = (const float*)d_in[8];
  const float*  bv = (const float*)d_in[9];
  const int8_t* wo = (const int8_t*)d_in[10];
  const float*  so = (const float*)d_in[11];
  const float*  bo = (const float*)d_in[12];

  // Workspace: q | k | v^T (f16, 8 MiB each) | y (f32, 16 MiB) = 40 MiB
  char* ws = (char*)d_ws;
  _Float16* qh  = (_Float16*)(ws);
  _Float16* kh  = (_Float16*)(ws + (size_t)8  * 1024 * 1024);
  _Float16* vth = (_Float16*)(ws + (size_t)16 * 1024 * 1024);
  float*    y   = (float*)   (ws + (size_t)24 * 1024 * 1024);

  const dim3 gg(Cdim / 64, Mrows / 128);   // (16, 32)
  gemm_i8<1><<<gg, 256, 0, stream>>>(x, wq, sq, bq, qh,  nullptr);
  gemm_i8<1><<<gg, 256, 0, stream>>>(x, wk, sk, bk, kh,  nullptr);
  gemm_i8<2><<<gg, 256, 0, stream>>>(x, wv, sv, bv, vth, nullptr);

  const dim3 ga(Tdim / 128, 2 * Hn);       // (16, 32)
  flash_attn<<<ga, 256, 0, stream>>>(qh, kh, vth, y);

  gemm_i8<0><<<gg, 256, 0, stream>>>(y, wo, so, bo, nullptr, (float*)d_out);
}
